// ExpertGating_74191265071206
// MI455X (gfx1250) — matrix operand order repaired and checked by reading, not compiled, not run on hardware
//
#include <hip/hip_runtime.h>
#include <hip/hip_bf16.h>

typedef __attribute__((ext_vector_type(16))) _Float16 v16h;
typedef __attribute__((ext_vector_type(8)))  _Float16 v8h;
typedef __attribute__((ext_vector_type(8)))  float    v8f;
typedef __attribute__((ext_vector_type(4)))  float    v4f;

#define NUM_TOKENS  16384
#define HIDDEN      4096
#define NUM_EXPERTS 64
#define TOP_K       8

#define KC          256          // K-chunk staged in LDS
#define NCHUNKS     (HIDDEN / KC)
#define WSTRIDE     72           // LDS weight row stride in halves (64 + 8 pad)
#define LSTRIDE     68           // LDS logit row stride in floats (64 + 4 pad)
#define WAVES       8            // waves per block
#define TOK_PER_BLK (WAVES * 16) // 128 tokens per block

#define WBUF_HALVES (KC * WSTRIDE)                 // one weight buffer, in halves
#define SMEM_BYTES  (2 * WBUF_HALVES * sizeof(_Float16) + WAVES * 16 * LSTRIDE * sizeof(float))

#ifndef __has_builtin
#define __has_builtin(x) 0
#endif
#if __has_builtin(__builtin_amdgcn_tensor_load_to_lds) && __has_builtin(__builtin_amdgcn_s_wait_tensorcnt)
#define HAVE_TDM 1
#else
#define HAVE_TDM 0
#endif

// ---------------------------------------------------------------------------
// One-time prep: gate_w (64,4096) f32 -> gwT (4096,64) f16 (K-major, contiguous
// per K-chunk so the TDM can stream it as a plain 2D tile).
// ---------------------------------------------------------------------------
__global__ void __launch_bounds__(256)
gw_transpose_f16_kernel(const float* __restrict__ gw, _Float16* __restrict__ gwT)
{
    const int idx = blockIdx.x * 256 + threadIdx.x;  // over 64*4096, coalesced read
    const int e = idx >> 12;                         // / HIDDEN
    const int k = idx & (HIDDEN - 1);
    gwT[(size_t)k * NUM_EXPERTS + e] = (_Float16)gw[idx];
}

#if HAVE_TDM
// Issue a TDM load of one K-chunk of pre-transposed f16 weights into LDS.
// Tile: tile_dim0 = 64 halves (one K row, 128B) x tile_dim1 = KC rows.
// TDM padding inserts 4 dwords (16B) after every 32 dwords (128B) stored,
// producing the [KC][WSTRIDE=72] layout in LDS for conflict-spread B reads.
__device__ __forceinline__ void tdm_load_weights(const _Float16* gwT, int k0, unsigned lds_off)
{
    typedef unsigned int v4u_ __attribute__((ext_vector_type(4)));
    typedef int          v8i_ __attribute__((ext_vector_type(8)));
    typedef int          v4i_ __attribute__((ext_vector_type(4)));

    const unsigned long long gaddr =
        (unsigned long long)(const void*)(gwT + (size_t)k0 * NUM_EXPERTS);

    v4u_ g0;
    g0[0] = 1u;                                                  // count=1, user D#
    g0[1] = lds_off;                                             // LDS byte address
    g0[2] = (unsigned)(gaddr & 0xFFFFFFFFu);                     // global_addr[31:0]
    g0[3] = (unsigned)((gaddr >> 32) & 0x1FFFFFFu) | (2u << 30); // [56:32] | type=2

    v8i_ g1;
    g1[0] = (1 << 16)        // data_size = 2 bytes
          | (1 << 20)        // pad_enable
          | (4 << 22)        // pad_interval code 4 -> 32 dwords (= 128B row)
          | (3 << 25);       // pad_amount  code 3 ->  4 dwords (= 16B)
    g1[1] = (NUM_EXPERTS & 0xFFFF) << 16;   // tensor_dim0[15:0] in bits [63:48]
    g1[2] = (HIDDEN & 0xFFFF) << 16;        // tensor_dim1[15:0] in bits [95:80]
    g1[3] = (NUM_EXPERTS & 0xFFFF) << 16;   // tile_dim0 in bits [127:112]
    g1[4] = KC;                             // tile_dim1 = KC rows; tile_dim2 = 0
    g1[5] = NUM_EXPERTS;                    // tensor_dim0_stride[31:0]
    g1[6] = 0;
    g1[7] = 0;

    v4i_ z4 = {0, 0, 0, 0};
#if __clang_major__ >= 23
    v8i_ z8 = {0, 0, 0, 0, 0, 0, 0, 0};
    __builtin_amdgcn_tensor_load_to_lds(g0, g1, z4, z4, z8, 0);
#else
    __builtin_amdgcn_tensor_load_to_lds(g0, g1, z4, z4, 0);
#endif
}
#endif // HAVE_TDM

__global__ void __launch_bounds__(256, 1)
expert_gating_kernel(const float* __restrict__ x,
                     const _Float16* __restrict__ gwT,
                     const float* __restrict__ gb,
                     float* __restrict__ out_w,
                     float* __restrict__ out_i)
{
    extern __shared__ char smem[];
    _Float16* wlds = (_Float16*)smem;                                       // 2 x [KC][WSTRIDE]
    float*    llds = (float*)(smem + 2 * WBUF_HALVES * sizeof(_Float16));   // [WAVES][16][LSTRIDE]

    const int tid  = threadIdx.x;
    const int wave = tid >> 5;
    const int lane = tid & 31;

    const int tokBase = blockIdx.x * TOK_PER_BLK + wave * 16;

    // A-operand addressing per the 16-bit 16x32 A layout:
    // lanes 0..15  hold row=lane,    K = {0..7, 16..23}
    // lanes 16..31 hold row=lane-16, K = {8..15, 24..31}
    const int arow  = lane & 15;
    const int khalf = (lane >> 4) * 8;
    const float* xrow = x + (size_t)(tokBase + arow) * HIDDEN;

    v8f acc[4];
    #pragma unroll
    for (int n = 0; n < 4; ++n) acc[n] = (v8f){};

#if HAVE_TDM
    const unsigned ldsw0 = (unsigned)(unsigned long long)(const void*)wlds;
    if (wave == 0) {
        tdm_load_weights(gwT, 0, ldsw0);
        __builtin_amdgcn_s_wait_tensorcnt(0);
    }
    __syncthreads();
#endif

    for (int c = 0; c < NCHUNKS; ++c) {
        const int k0 = c * KC;

#if HAVE_TDM
        // Kick off the DMA for the next chunk into the other buffer, then
        // overlap it with this chunk's WMMAs.
        if (wave == 0 && c + 1 < NCHUNKS)
            tdm_load_weights(gwT, k0 + KC,
                             ldsw0 + (unsigned)(((c + 1) & 1) * WBUF_HALVES * sizeof(_Float16)));
        const _Float16* wbase = wlds + (size_t)(c & 1) * WBUF_HALVES;
#else
        __syncthreads();
        // cooperative padded copy of the f16 chunk (fallback when no TDM builtin)
        for (int j = tid; j < (KC * NUM_EXPERTS) / 8; j += 256) {
            const int kr = j >> 3, seg = j & 7;
            *(v8h*)(wlds + (size_t)kr * WSTRIDE + seg * 8) =
                *(const v8h*)(gwT + (size_t)(k0 + kr) * NUM_EXPERTS + seg * 8);
        }
        __syncthreads();
        const _Float16* wbase = wlds;
#endif

        // speculative prefetch of next x chunk (global_prefetch_b8)
        if (k0 + KC < HIDDEN) __builtin_prefetch(xrow + k0 + KC, 0, 0);

        const float* xc = xrow + k0;

        auto loadA = [&](int kk) -> v16h {
            const float* ap = xc + kk + khalf;
            v4f a0 = *(const v4f*)(ap + 0);
            v4f a1 = *(const v4f*)(ap + 4);
            v4f a2 = *(const v4f*)(ap + 16);
            v4f a3 = *(const v4f*)(ap + 20);
            v16h A;
            #pragma unroll
            for (int i = 0; i < 4; ++i) {
                A[i]      = (_Float16)a0[i];
                A[4 + i]  = (_Float16)a1[i];
                A[8 + i]  = (_Float16)a2[i];
                A[12 + i] = (_Float16)a3[i];
            }
            return A;
        };
        auto loadB = [&](int kk, int n) -> v16h {
            v16h B;
            #pragma unroll
            for (int s = 0; s < 16; ++s)
                B[s] = wbase[(size_t)(kk + khalf + (s & 7) + (s >> 3) * 16) * WSTRIDE + n * 16 + (lane & 15)];
            return B;
        };

        // ---- GEMM over this K-chunk: software-pipelined 8 steps of K=32.
        // Next step's A (global+cvt) and B (LDS) land in fresh registers while
        // this step's four WMMAs execute: no dscnt-0 stall per WMMA, no WAR
        // v_nops on the A registers.
        v16h A  = loadA(0);
        v16h B0 = loadB(0, 0), B1 = loadB(0, 1), B2 = loadB(0, 2), B3 = loadB(0, 3);
        #pragma unroll
        for (int kk = 0; kk < KC; kk += 32) {
            v16h An = A, B0n = B0, B1n = B1, B2n = B2, B3n = B3;
            if (kk + 32 < KC) {
                An  = loadA(kk + 32);
                B0n = loadB(kk + 32, 0);
                B1n = loadB(kk + 32, 1);
                B2n = loadB(kk + 32, 2);
                B3n = loadB(kk + 32, 3);
            }
            acc[0] = __builtin_amdgcn_wmma_f32_16x16x32_f16(false, A, false, B0, (short)0, acc[0], false, false);
            acc[1] = __builtin_amdgcn_wmma_f32_16x16x32_f16(false, A, false, B1, (short)0, acc[1], false, false);
            acc[2] = __builtin_amdgcn_wmma_f32_16x16x32_f16(false, A, false, B2, (short)0, acc[2], false, false);
            acc[3] = __builtin_amdgcn_wmma_f32_16x16x32_f16(false, A, false, B3, (short)0, acc[3], false, false);
            A = An; B0 = B0n; B1 = B1n; B2 = B2n; B3 = B3n;
        }

#if HAVE_TDM
        __syncthreads();   // all waves finished reading buf[c&1] before it is reused
        if (wave == 0 && c + 1 < NCHUNKS) __builtin_amdgcn_s_wait_tensorcnt(0);
        __syncthreads();   // next chunk's buffer now valid for every wave
#endif
    }

    // ---- Add bias, spill logits to LDS (per-wave 16x64 tile) ----
    // C layout: VGPR j, lanes 0..15 -> M=j, N=lane ; lanes 16..31 -> M=j+8, N=lane-16
    float* myl = llds + (size_t)wave * 16 * LSTRIDE;
    const int mbase = (lane >> 4) * 8;
    #pragma unroll
    for (int n = 0; n < 4; ++n) {
        const int e = n * 16 + (lane & 15);
        const float b = gb[e];
        #pragma unroll
        for (int j = 0; j < 8; ++j)
            myl[(size_t)(mbase + j) * LSTRIDE + e] = acc[n][j] + b;
    }
    __syncthreads();

    // ---- Per-token top-8 + softmax.
    // Lane pair (t, t+16) splits the 64 experts 32/32: each half-lane finds a
    // local top-8, candidates are exchanged with __shfl_xor(16), and an 8-of-16
    // selection merges. Lanes 0..15 (low-expert half first in candidate order,
    // so strict '>' keeps the lowest expert index on ties) write the result.
    {
        const int tokLane = lane & 15;
        const int eoff    = (lane >> 4) * 32;          // 0 or 32
        const float* row  = myl + (size_t)tokLane * LSTRIDE + eoff;

        float v[32];
        #pragma unroll
        for (int e = 0; e < 32; ++e) v[e] = row[e];

        float lv[TOP_K]; int li[TOP_K];
        #pragma unroll
        for (int k = 0; k < TOP_K; ++k) {
            float m = -__builtin_inff(); int mi = 0;
            #pragma unroll
            for (int e = 0; e < 32; ++e)
                if (v[e] > m) { m = v[e]; mi = e; }
            lv[k] = m; li[k] = mi + eoff;
            #pragma unroll
            for (int e = 0; e < 32; ++e)
                v[e] = (e == mi) ? -__builtin_inff() : v[e];
        }

        // exchange the partner half-lane's candidate list (wave32 permute)
        float cv[2 * TOP_K]; int ci[2 * TOP_K];
        #pragma unroll
        for (int k = 0; k < TOP_K; ++k) {
            cv[k] = lv[k];
            ci[k] = li[k];
            cv[TOP_K + k] = __shfl_xor(lv[k], 16, 32);
            ci[TOP_K + k] = __shfl_xor(li[k], 16, 32);
        }

        float tv[TOP_K]; int ti[TOP_K];
        #pragma unroll
        for (int k = 0; k < TOP_K; ++k) {
            float m = -__builtin_inff(); int mi = 0, mslot = 0;
            #pragma unroll
            for (int e = 0; e < 2 * TOP_K; ++e)
                if (cv[e] > m) { m = cv[e]; mi = ci[e]; mslot = e; }
            tv[k] = m; ti[k] = mi;
            #pragma unroll
            for (int e = 0; e < 2 * TOP_K; ++e)
                cv[e] = (e == mslot) ? -__builtin_inff() : cv[e];
        }

        if (lane < 16) {
            // softmax over the 8 selected logits (tv[0] is the max: descending)
            const float mx = tv[0];
            float ew[TOP_K];
            float s = 0.0f;
            #pragma unroll
            for (int k = 0; k < TOP_K; ++k) { ew[k] = __expf(tv[k] - mx); s += ew[k]; }
            const float inv = __frcp_rn(s);

            const size_t tok = (size_t)(tokBase + lane);
            #pragma unroll
            for (int k = 0; k < TOP_K; ++k) {
                out_w[tok * TOP_K + k] = ew[k] * inv;
                out_i[tok * TOP_K + k] = (float)ti[k];
            }
        }
    }
}

extern "C" void kernel_launch(void* const* d_in, const int* in_sizes, int n_in,
                              void* d_out, int out_size, void* d_ws, size_t ws_size,
                              hipStream_t stream) {
    const float* x  = (const float*)d_in[0];   // (16384, 4096) f32
    const float* gw = (const float*)d_in[1];   // (64, 4096)    f32
    const float* gb = (const float*)d_in[2];   // (64,)         f32
    float* out_w = (float*)d_out;                                   // (16384, 8)
    float* out_i = (float*)d_out + (size_t)NUM_TOKENS * TOP_K;      // (16384, 8) indices

    _Float16* gwT = (_Float16*)d_ws;           // (4096, 64) f16, 512 KB scratch

    // prep: transpose+convert gate_w once per launch (0.2% of total traffic)
    gw_transpose_f16_kernel<<<dim3((NUM_EXPERTS * HIDDEN) / 256), dim3(256), 0, stream>>>(gw, gwT);

    dim3 grid(NUM_TOKENS / TOK_PER_BLK);   // 128 blocks
    dim3 block(256);                       // 8 waves (wave32)
    expert_gating_kernel<<<grid, block, SMEM_BYTES, stream>>>(x, gwT, gb, out_w, out_i);
}
